// CannyFilter_84404697301059
// MI455X (gfx1250) — matrix operand order repaired and checked by reading, not compiled, not run on hardware
//
#include <hip/hip_runtime.h>
#include <cstdint>

// ---------------------------------------------------------------------------
// Canny edge filter for MI455X (gfx1250), faithful to the JAX reference
// (including its cross-batch flat-gather in the NMS stage, B == 8).
//
// Kernel A: gaussian blur (horizontal scalar f32->f16 transposed, vertical via
//           v_wmma_f32_16x16x32_f16 with contiguous B fragments) + sobel +
//           |grad| + orientation index; image tile staged via async
//           global->LDS copies (ASYNCcnt).
// Kernel B: NMS (cross-batch gather, L2-resident mag) + hysteresis + borders.
// ---------------------------------------------------------------------------

typedef __attribute__((ext_vector_type(16))) _Float16 v16h;
typedef __attribute__((ext_vector_type(8)))  float    v8f;

#define IMG_H   1024
#define IMG_W   1024
#define HW      (1024 * 1024)
#define NBATCH  8
#define NCH     3
#define TILE    32

// gaussian window, STD=1, FILTER_SIZE=5, unnormalized (scipy gaussian)
__device__ __constant__ float G5[5] = {
    0.13533528323661270f, 0.60653065971263342f, 1.0f,
    0.60653065971263342f, 0.13533528323661270f };

// directional NMS offsets (dy,dx), index d = filter index in reference dirf
__device__ __constant__ int OFF_DY[8] = { 0, 1, 1,  1,  0, -1, -1, -1 };
__device__ __constant__ int OFF_DX[8] = { 1, 1, 0, -1, -1, -1,  0,  1 };

// ---------------------------------------------------------------------------
// async global -> LDS copy (gfx1250 GLOBAL_LOAD_ASYNC_TO_LDS_B32, ASYNCcnt)
// ---------------------------------------------------------------------------
#if defined(__has_builtin)
#if __has_builtin(__builtin_amdgcn_global_load_async_to_lds_b32) && \
    __has_builtin(__builtin_amdgcn_s_wait_asynccnt)
#define USE_ASYNC_LDS 1
#endif
#endif

__device__ __forceinline__ void async_copy_f32(const float* g, float* l) {
#ifdef USE_ASYNC_LDS
    float* gnc = const_cast<float*>(g);
    __builtin_amdgcn_global_load_async_to_lds_b32(
        (__attribute__((address_space(1))) int*)(gnc),
        (__attribute__((address_space(3))) int*)(l),
        0, 0);
#else
    *l = *g;
#endif
}

__device__ __forceinline__ void async_wait_all() {
#ifdef USE_ASYNC_LDS
    __builtin_amdgcn_s_wait_asynccnt(0);
#endif
}

// ---------------------------------------------------------------------------
// Kernel A geometry (coords relative to the 32x32 tile):
//   img    : y,x in [-3,34]  (38x38, row stride 40 f32)  per channel
//   hblurT : TRANSPOSED f16 [ch][col 0..47][row 0..71], row stride 72
//            valid data: col = x+1 for x in [-1,32], row = y+3, y in [-3,34]
//            everything else zero (conv zero padding / WMMA K-tail)
//   blur   : y,x in [-1,46]  (48x48 f32)                 per channel
// blur rows for WMMA: out row (buf) = 16*rg + m <- hblurT rows 16*rg + m..m+4
// ---------------------------------------------------------------------------
#define IMG_R    38
#define IMG_S    40     // img row stride (f32)
#define HBT_C    48     // hblurT columns
#define HBT_S    72     // hblurT row stride (halves); max row used = 63
#define BL_R     48
#define BL_W     48

#define HBT_BYTES  (NCH * HBT_C * HBT_S * 2)       // 20736
#define IMG_BYTES  (NCH * IMG_R * IMG_S * 4)       // 18240
#define BL_BYTES   (NCH * BL_R * BL_W * 4)         // 27648
// img region is dead after the horizontal pass; blur aliases it.
#define SMEM_BYTES (HBT_BYTES + BL_BYTES)          // 48384

__global__ __launch_bounds__(256)
void canny_mag_kernel(const float* __restrict__ img,
                      float* __restrict__ magW,
                      unsigned char* __restrict__ idxW)
{
    __shared__ __align__(32) unsigned char s_raw[SMEM_BYTES];
    _Float16* s_hblur = (_Float16*)(s_raw);                 // persistent
    float*    s_img   = (float*)(s_raw + HBT_BYTES);        // phase 1
    float*    s_blur  = (float*)(s_raw + HBT_BYTES);        // phase 2 (alias)

    const int tid = threadIdx.x;
    const int X0  = blockIdx.x * TILE;
    const int Y0  = blockIdx.y * TILE;
    const int bz  = blockIdx.z;

    // ---- stage image tile (halo 3) into LDS with async copies -------------
    const int NIMG = NCH * IMG_R * IMG_R;
    for (int i = tid; i < NIMG; i += 256) {
        int c = i / (IMG_R * IMG_R);
        int r = i - c * (IMG_R * IMG_R);
        int y = r / IMG_R, x = r - y * IMG_R;
        int gy = Y0 - 3 + y, gx = X0 - 3 + x;
        if (gy >= 0 && gy < IMG_H && gx >= 0 && gx < IMG_W) {
            const float* gp = img + (((size_t)(bz * NCH + c) * IMG_H + gy) * IMG_W + gx);
            async_copy_f32(gp, &s_img[(c * IMG_R + y) * IMG_S + x]);
        }
    }
    // zero whole hblurT buffer (zero padding + WMMA K tail) while loads fly
    {
        uint32_t* hz = (uint32_t*)s_hblur;
        for (int i = tid; i < (HBT_BYTES / 4); i += 256) hz[i] = 0u;
    }

    async_wait_all();
    __syncthreads();

    // zero-fill image-OOB entries (conv zero padding)
    for (int i = tid; i < NIMG; i += 256) {
        int c = i / (IMG_R * IMG_R);
        int r = i - c * (IMG_R * IMG_R);
        int y = r / IMG_R, x = r - y * IMG_R;
        int gy = Y0 - 3 + y, gx = X0 - 3 + x;
        if (!(gy >= 0 && gy < IMG_H && gx >= 0 && gx < IMG_W))
            s_img[(c * IMG_R + y) * IMG_S + x] = 0.0f;
    }
    __syncthreads();

    // ---- horizontal 5-tap gaussian, store transposed f16 ------------------
    // hblur col x (0..33) <-> img cols x..x+4 ; stored at [c][x][y]
    const int NHB = NCH * IMG_R * 34;
    for (int i = tid; i < NHB; i += 256) {
        int c = i / (IMG_R * 34);
        int r = i - c * (IMG_R * 34);
        int y = r / 34, x = r - y * 34;
        const float* ip = &s_img[(c * IMG_R + y) * IMG_S + x];
        float v = G5[0] * ip[0] + G5[1] * ip[1] + G5[2] * ip[2]
                + G5[3] * ip[3] + G5[4] * ip[4];
        s_hblur[(c * HBT_C + x) * HBT_S + y] = (_Float16)v;
    }
    __syncthreads();

    // ---- vertical 5-tap gaussian via WMMA f16 (K=32 holds 20-row band) ----
    const int lane = tid & 31;
    const int wave = tid >> 5;
    const int M    = lane & 15;
    const bool hi  = lane >= 16;

    v16h A;
    #pragma unroll
    for (int h = 0; h < 16; ++h) {
        int k = (hi ? 16 : 0) + h;
        int d = k - M;
        A[h] = (d >= 0 && d < 5) ? (_Float16)G5[d] : (_Float16)0.0f;
    }
    const int kbase = hi ? 16 : 0;

    for (int job = wave; job < 27; job += 8) {   // 3 ch x 3 rowgrp x 3 colgrp
        int c  = job / 9;
        int r9 = job - c * 9;
        int rg = r9 / 3, cg = r9 - rg * 3;
        int col = 16 * cg + M;

        // contiguous, 16B-aligned K-slice: rows 16*rg+kbase .. +15
        const v16h* bp = reinterpret_cast<const v16h*>(
            &s_hblur[(c * HBT_C + col) * HBT_S + 16 * rg + kbase]);
        v16h Bm = *bp;

        v8f C = {};
        v8f D = __builtin_amdgcn_wmma_f32_16x16x32_f16(
                    false, A, false, Bm, (short)0, C, false, false);

        int rowbase = 16 * rg + (hi ? 8 : 0);
        #pragma unroll
        for (int r = 0; r < 8; ++r)
            s_blur[(c * BL_R + rowbase + r) * BL_W + col] = D[r];
    }
    __syncthreads();

    // ---- zero blur entries outside the image (sobel zero padding) ---------
    for (int i = tid; i < NCH * 34 * 34; i += 256) {
        int c = i / (34 * 34);
        int r = i - c * (34 * 34);
        int y = r / 34, x = r - y * 34;         // buffer rows/cols 0..33
        int gy = Y0 - 1 + y, gx = X0 - 1 + x;
        if (!(gy >= 0 && gy < IMG_H && gx >= 0 && gx < IMG_W))
            s_blur[(c * BL_R + y) * BL_W + x] = 0.0f;
    }
    __syncthreads();

    // ---- sobel + magnitude + orientation index ----------------------------
    for (int i = tid; i < TILE * TILE; i += 256) {
        int py = i >> 5, px = i & 31;
        int row = py + 1, col = px + 1;
        float mag = 0.0f, ang = 0.0f;
        #pragma unroll
        for (int c = 0; c < NCH; ++c) {
            const float* bp = &s_blur[(c * BL_R + row) * BL_W + col];
            float a00 = bp[-BL_W - 1], a01 = bp[-BL_W], a02 = bp[-BL_W + 1];
            float a10 = bp[-1],                          a12 = bp[1];
            float a20 = bp[ BL_W - 1], a21 = bp[ BL_W], a22 = bp[ BL_W + 1];
            float gxv = (a00 - a02) + 2.0f * (a10 - a12) + (a20 - a22);
            float gyv = (a00 - a20) + 2.0f * (a01 - a21) + (a02 - a22);
            mag += sqrtf(gxv * gxv + gyv * gyv);
            ang += atan2f(gyv, gxv);
        }
        // deg/45 = ang*(4/pi) + 4 ; idx_pos = round(.) mod 8
        float q45 = rintf(ang * 1.27323954473516269f + 4.0f);
        int q = (int)q45;
        int idxp = ((q % 8) + 8) & 7;

        size_t gp = ((size_t)bz << 20) + (size_t)(Y0 + py) * IMG_W + (X0 + px);
        magW[gp] = mag;
        idxW[gp] = (unsigned char)idxp;
    }
}

// ---------------------------------------------------------------------------
// Kernel B: NMS (faithful cross-batch gather: positive[b] reads batch
// idx_pos with direction offset = b) + hysteresis + border zeroing.
// mag array (32 MB) is L2-resident on MI455X (192 MB L2).
// ---------------------------------------------------------------------------
#define THIN_S 36

__global__ __launch_bounds__(256)
void canny_nms_hyst_kernel(const float* __restrict__ magW,
                           const unsigned char* __restrict__ idxW,
                           float* __restrict__ out)
{
    __shared__ float s_thin[34 * THIN_S];

    const int tid = threadIdx.x;
    const int X0  = blockIdx.x * TILE;
    const int Y0  = blockIdx.y * TILE;
    const int bz  = blockIdx.z;

    const int dy = OFF_DY[bz];     // direction index == batch index (B==8)
    const int dx = OFF_DX[bz];

    // thin over tile + halo 1
    for (int i = tid; i < 34 * 34; i += 256) {
        int y = i / 34, x = i - (i / 34) * 34;
        int gy = Y0 - 1 + y, gx = X0 - 1 + x;
        float t = 0.0f;
        if (gy >= 0 && gy < IMG_H && gx >= 0 && gx < IMG_W) {
            size_t q = (size_t)gy * IMG_W + gx;
            float tb = magW[((size_t)bz << 20) + q];
            int  ip  = idxW[((size_t)bz << 20) + q];
            int  in_ = (ip + 4) & 7;
            int  ny = gy + dy, nx = gx + dx;
            bool nin = (ny >= 0 && ny < IMG_H && nx >= 0 && nx < IMG_W);
            size_t qn = nin ? ((size_t)ny * IMG_W + nx) : 0;
            float mp  = magW[((size_t)ip  << 20) + q];
            float mn  = magW[((size_t)in_ << 20) + q];
            float mpn = nin ? magW[((size_t)ip  << 20) + qn] : 0.0f;
            float mnn = nin ? magW[((size_t)in_ << 20) + qn] : 0.0f;
            float pos = mp - mpn;
            float neg = mn - mnn;
            t = (fminf(pos, neg) > 0.0f) ? tb : 0.0f;
        }
        s_thin[y * THIN_S + x] = t;
    }
    __syncthreads();

    // hysteresis + borders
    for (int i = tid; i < TILE * TILE; i += 256) {
        int py = i >> 5, px = i & 31;
        int y = py + 1, x = px + 1;
        const float* tp = &s_thin[y * THIN_S + x];
        float t = tp[0];
        bool nb = (tp[-THIN_S - 1] > 100.0f) || (tp[-THIN_S] > 100.0f) ||
                  (tp[-THIN_S + 1] > 100.0f) || (tp[-1]      > 100.0f) ||
                  (tp[1]           > 100.0f) || (tp[ THIN_S - 1] > 100.0f) ||
                  (tp[ THIN_S]     > 100.0f) || (tp[ THIN_S + 1] > 100.0f);
        bool edge = (t > 100.0f) || ((t >= 10.0f) && (t <= 100.0f) && nb);
        int gy = Y0 + py, gx = X0 + px;
        if (gy == 0 || gy == IMG_H - 1 || gx == 0 || gx == IMG_W - 1)
            edge = false;
        out[((size_t)bz << 20) + (size_t)gy * IMG_W + gx] = edge ? 1.0f : 0.0f;
    }
}

// ---------------------------------------------------------------------------
extern "C" void kernel_launch(void* const* d_in, const int* in_sizes, int n_in,
                              void* d_out, int out_size, void* d_ws, size_t ws_size,
                              hipStream_t stream)
{
    const float* img = (const float*)d_in[0];
    float* magW = (float*)d_ws;                                   // 32 MB
    unsigned char* idxW = (unsigned char*)d_ws
                        + (size_t)NBATCH * HW * sizeof(float);    // +8 MB

    dim3 grid(IMG_W / TILE, IMG_H / TILE, NBATCH);
    dim3 block(256);
    canny_mag_kernel<<<grid, block, 0, stream>>>(img, magW, idxW);
    canny_nms_hyst_kernel<<<grid, block, 0, stream>>>(magW, idxW, (float*)d_out);
}
